// WindowAttention_5634997092543
// MI455X (gfx1250) — matrix operand order repaired and checked by reading, not compiled, not run on hardware
//
#include <hip/hip_runtime.h>
#include <hip/hip_bf16.h>

typedef unsigned short u16;
typedef __attribute__((ext_vector_type(16))) __bf16 v16bf;
typedef __attribute__((ext_vector_type(8)))  float  v8f;
typedef __attribute__((ext_vector_type(4)))  unsigned int u32x4;
typedef __attribute__((ext_vector_type(8)))  unsigned int u32x8;

struct alignas(16) U128 { unsigned int w[4]; };
struct alignas(32) U256 { U128 lo, hi; };
struct alignas(8)  U64  { unsigned int w[2]; };

#define DIM 192
#define NH 6
#define HD 32
#define NTOK 64
#define NWIN 4096
#define XP 200   // Xb pitch (bf16 elems), mult of 8, bank-stagger
#define QP 40
#define KP 72
#define VP 40
#define SP 68    // f32 pitch, mult of 4 -> 16B-aligned rows
#define PP 72
#define OPI 200

// ---- LDS arena layout (bytes); Xs aliases Sf/Ph/Ob (prologue-only use) ----
#define XB_OFF 0        // u16 [64*200] = 25600
#define QH_OFF 25600    // u16 [64*40]  = 5120
#define KT_OFF 30720    // u16 [32*72]  = 4608
#define VH_OFF 35328    // u16 [64*40]  = 5120
#define MB_OFF 40448    // f32 [64*64]  = 16384  (mask tile, reused by all heads)
#define SF_OFF 56832    // f32 [64*68]  = 17408
#define PH_OFF 74240    // u16 [64*72]  = 9216
#define OB_OFF 83456    // u16 [64*200] = 25600
#define XS_OFF 56832    // f32 [64*192] = 49152  (aliases SF..OB region)
#define SMEM_BYTES 109056

__device__ __forceinline__ u16 f2bf(float f) {
  unsigned u = __float_as_uint(f);
  unsigned r = u + 0x7FFFu + ((u >> 16) & 1u);   // round-to-nearest-even
  return (u16)(r >> 16);
}

__device__ __forceinline__ unsigned f2bf2(float lo, float hi) {
  return (unsigned)f2bf(lo) | ((unsigned)f2bf(hi) << 16);
}

__device__ __forceinline__ v16bf frag2(const u16* p0, const u16* p1) {
  U256 t; t.lo = *(const U128*)p0; t.hi = *(const U128*)p1;
  return __builtin_bit_cast(v16bf, t);
}

// A fragment (16x32 bf16), row-major buffer, pitch in elements.
// lanes 0-15 = M0-15 / K {0..7,16..23}; lanes 16-31 = M0-15 / K {8..15,24..31}
__device__ __forceinline__ v16bf afrag(const u16* buf, int pitch, int mbase, int kbase, int lane) {
  int m  = mbase + (lane & 15);
  int kh = (lane >> 4) << 3;
  const u16* p = buf + m * pitch + kbase + kh;
  return frag2(p, p + 16);
}

// B fragment (32x16 bf16), row-major [K][N]: lanes 0-15 = N0-15 / K {0..7,16..23}; lanes 16-31 = N0-15 / K {8..15,24..31}
__device__ __forceinline__ v16bf bfrag(const u16* buf, int pitch, int kbase, int nbase, int lane) {
  const u16* p = buf + (kbase + ((lane >> 4) << 3)) * pitch + nbase + (lane & 15);
  u16 t[16];
  #pragma unroll
  for (int s = 0; s < 8; ++s) { t[s] = p[s * pitch]; t[s + 8] = p[(s + 16) * pitch]; }
  return __builtin_bit_cast(v16bf, t);
}

// ---- Tensor Data Mover: 2D f32 tile load global -> LDS (D# per ISA 08 §8.3/8.4) ----
__device__ __forceinline__ void tdm_load_2d_f32(unsigned lds_byte_addr, const void* gaddr,
                                                unsigned width_elems, unsigned height,
                                                unsigned stride_elems) {
  unsigned long long ga = (unsigned long long)gaddr;
  u32x4 g0;
  g0[0] = 1u;                                          // count=1, user descriptor
  g0[1] = lds_byte_addr;                               // lds_addr
  g0[2] = (unsigned)(ga & 0xFFFFFFFFull);              // global_addr[31:0]
  g0[3] = (unsigned)((ga >> 32) & 0x01FFFFFFull)       // global_addr[56:32]
        | 0x80000000u;                                 // type = 2 ("image")
  u32x8 g1;
  g1[0] = (2u << 16);                                  // wg_mask=0, data_size=2 (4 bytes)
  g1[1] = (width_elems & 0xFFFFu) << 16;               // tensor_dim0[15:0]
  g1[2] = (width_elems >> 16) | ((height & 0xFFFFu) << 16);   // dim0 hi | tensor_dim1 lo
  g1[3] = (height >> 16) | ((width_elems & 0xFFFFu) << 16);   // dim1 hi | tile_dim0
  g1[4] = (height & 0xFFFFu);                          // tile_dim1 (tile_dim2=0)
  g1[5] = stride_elems;                                // tensor_dim0_stride[31:0]
  g1[6] = 0u;                                          // stride hi | tensor_dim1_stride lo
  g1[7] = 0u;
  asm volatile("tensor_load_to_lds %0, %1" :: "s"(g0), "s"(g1) : "memory");
}

__global__ void prep_rpb_kernel(const float* __restrict__ bt, const int* __restrict__ ri,
                                float* __restrict__ rpb) {
  int idx = blockIdx.x * 256 + threadIdx.x;
  if (idx >= NH * NTOK * NTOK) return;
  int h  = idx >> 12;
  int ij = idx & 4095;
  rpb[idx] = bt[ri[ij] * NH + h];
}

// out[k*Nn + n] = bf16(w[n*K + k])  (transpose to [K][N] for B-fragment loads)
__global__ void prep_w_kernel(const float* __restrict__ w, u16* __restrict__ o, int K, int Nn) {
  int idx = blockIdx.x * 256 + threadIdx.x;
  if (idx >= K * Nn) return;
  int k = idx / Nn, n = idx - k * Nn;
  o[idx] = f2bf(w[n * K + k]);
}

__global__ __launch_bounds__(256)
void winattn_kernel(const float* __restrict__ x, const float* __restrict__ mask,
                    const u16* __restrict__ WqkvB, const float* __restrict__ qkv_b,
                    const u16* __restrict__ ProjB, const float* __restrict__ proj_b,
                    const float* __restrict__ rpb, float* __restrict__ out) {
  __shared__ alignas(16) char smem[SMEM_BYTES];
  u16*   Xb  = (u16*)(smem + XB_OFF);
  u16*   Qh  = (u16*)(smem + QH_OFF);
  u16*   KTh = (u16*)(smem + KT_OFF);
  u16*   Vh  = (u16*)(smem + VH_OFF);
  float* Mb  = (float*)(smem + MB_OFF);
  float* Sf  = (float*)(smem + SF_OFF);
  u16*   Ph  = (u16*)(smem + PH_OFF);
  u16*   Ob  = (u16*)(smem + OB_OFF);
  float* Xs  = (float*)(smem + XS_OFF);

  const int tid  = threadIdx.x;
  const int wid  = tid >> 5;
  const int lane = tid & 31;
  const int b    = blockIdx.x;
  const long long xbase = (long long)b * (NTOK * DIM);
  const float* xp = x + xbase;
  const float* mk = mask + (long long)(b & (NWIN - 1)) * (NTOK * NTOK);

  // ---- TDM-stage x tile (48KB f32) and mask tile (16KB f32) into LDS ----
  if (wid == 0) {
    tdm_load_2d_f32((unsigned)(unsigned long long)(void*)Xs, xp, DIM, NTOK, DIM);
    tdm_load_2d_f32((unsigned)(unsigned long long)(void*)Mb, mk, NTOK, NTOK, NTOK);
    __builtin_amdgcn_s_wait_tensorcnt(0);
  }
  __syncthreads();

  // convert staged x (f32, contiguous) -> Xb (bf16, pitched); b128 in, b64 out
  #pragma unroll
  for (int i4 = tid * 4; i4 < NTOK * DIM; i4 += 1024) {
    U128 q = *(const U128*)(Xs + i4);
    float f0 = __uint_as_float(q.w[0]), f1 = __uint_as_float(q.w[1]);
    float f2 = __uint_as_float(q.w[2]), f3 = __uint_as_float(q.w[3]);
    int r = i4 / DIM, c = i4 - r * DIM;
    U64 o; o.w[0] = f2bf2(f0, f1); o.w[1] = f2bf2(f2, f3);
    *(U64*)(Xb + r * XP + c) = o;
  }
  __syncthreads();

  for (int h = 0; h < NH; ++h) {
    // ---- QKV chunk for head h: 24 16x16 tiles (q,k,v x 4mt x 2nt), 3 per wave ----
    for (int s = 0; s < 3; ++s) {
      int tt = wid + (s << 3);
      int outsel = tt >> 3;          // 0=q 1=k 2=v
      int mt = (tt & 7) >> 1;
      int nt = tt & 1;
      int colg = outsel * DIM + h * HD + nt * 16;
      v8f acc = {};
      #pragma unroll
      for (int kt = 0; kt < 6; ++kt) {
        v16bf a  = afrag(Xb, XP, mt * 16, kt * 32, lane);
        v16bf bb = bfrag(WqkvB, 3 * DIM, kt * 32, colg, lane);
        acc = __builtin_amdgcn_wmma_f32_16x16x32_bf16(false, a, false, bb, (short)0, acc, false, false);
      }
      int nl = lane & 15;
      int mh = (lane >> 4) << 3;
      float bias = qkv_b[colg + nl];
      if (outsel == 0) {
        #pragma unroll
        for (int r = 0; r < 8; ++r)
          Qh[(mt * 16 + mh + r) * QP + nt * 16 + nl] = f2bf((acc[r] + bias) * 0.17677669529663687f);
      } else if (outsel == 1) {
        #pragma unroll
        for (int r = 0; r < 8; ++r)   // transposed store: k^T[d][token]
          KTh[(nt * 16 + nl) * KP + (mt * 16 + mh + r)] = f2bf(acc[r] + bias);
      } else {
        #pragma unroll
        for (int r = 0; r < 8; ++r)
          Vh[(mt * 16 + mh + r) * VP + nt * 16 + nl] = f2bf(acc[r] + bias);
      }
    }
    __syncthreads();

    // ---- S = Qh(64x32) @ KTh(32x64), K=32 -> one WMMA per tile; raw scores to LDS ----
    for (int s = 0; s < 2; ++s) {
      int tt = wid + (s << 3);
      int mt = tt >> 2, nt = tt & 3;
      v16bf a  = afrag(Qh, QP, mt * 16, 0, lane);
      v16bf bb = bfrag(KTh, KP, 0, nt * 16, lane);
      v8f acc = {};
      acc = __builtin_amdgcn_wmma_f32_16x16x32_bf16(false, a, false, bb, (short)0, acc, false, false);
      int nl = lane & 15, mh = (lane >> 4) << 3;
      #pragma unroll
      for (int r = 0; r < 8; ++r)
        Sf[(mt * 16 + mh + r) * SP + nt * 16 + nl] = acc[r];
    }
    __syncthreads();

    // ---- softmax (+rpb +mask, row-contiguous -> b128 loads), packed bf16 prob stores ----
    {
      const float* rp = rpb + h * (NTOK * NTOK);
      int row = tid >> 2, sub = tid & 3;
      int c0 = sub * 16;
      const float* sfr = Sf + row * SP + c0;
      const float* rpr = rp + row * NTOK + c0;
      const float* mbr = Mb + row * NTOK + c0;
      float vv[16];
      float mx = -3.4e38f;
      #pragma unroll
      for (int j = 0; j < 16; ++j) { vv[j] = sfr[j] + rpr[j] + mbr[j]; mx = fmaxf(mx, vv[j]); }
      mx = fmaxf(mx, __shfl_xor(mx, 1, 4));
      mx = fmaxf(mx, __shfl_xor(mx, 2, 4));
      float sum = 0.f;
      #pragma unroll
      for (int j = 0; j < 16; ++j) { vv[j] = __expf(vv[j] - mx); sum += vv[j]; }
      sum += __shfl_xor(sum, 1, 4);
      sum += __shfl_xor(sum, 2, 4);
      float inv = 1.f / sum;
      U128 p0, p1;
      #pragma unroll
      for (int j = 0; j < 4; ++j) p0.w[j] = f2bf2(vv[2 * j] * inv, vv[2 * j + 1] * inv);
      #pragma unroll
      for (int j = 0; j < 4; ++j) p1.w[j] = f2bf2(vv[8 + 2 * j] * inv, vv[9 + 2 * j] * inv);
      *(U128*)(Ph + row * PP + c0)     = p0;
      *(U128*)(Ph + row * PP + c0 + 8) = p1;
    }
    __syncthreads();

    // ---- O_h = Ph(64x64) @ Vh(64x32): 8 tiles, one per wave, K=64 ----
    {
      int mt = wid >> 1, nt = wid & 1;
      v8f acc = {};
      #pragma unroll
      for (int kt = 0; kt < 2; ++kt) {
        v16bf a  = afrag(Ph, PP, mt * 16, kt * 32, lane);
        v16bf bb = bfrag(Vh, VP, kt * 32, nt * 16, lane);
        acc = __builtin_amdgcn_wmma_f32_16x16x32_bf16(false, a, false, bb, (short)0, acc, false, false);
      }
      int nl = lane & 15, mh = (lane >> 4) << 3;
      #pragma unroll
      for (int r = 0; r < 8; ++r)
        Ob[(mt * 16 + mh + r) * OPI + h * HD + nt * 16 + nl] = f2bf(acc[r]);
    }
    __syncthreads();
  }

  // ---- projection: Y(64x192) = Ob @ ProjB + proj_b, store f32 to HBM ----
  float* yo = out + xbase;
  for (int tt = wid; tt < 48; tt += 8) {
    int mt = tt / 12, nt = tt - mt * 12;
    v8f acc = {};
    #pragma unroll
    for (int kt = 0; kt < 6; ++kt) {
      v16bf a  = afrag(Ob, OPI, mt * 16, kt * 32, lane);
      v16bf bb = bfrag(ProjB, DIM, kt * 32, nt * 16, lane);
      acc = __builtin_amdgcn_wmma_f32_16x16x32_bf16(false, a, false, bb, (short)0, acc, false, false);
    }
    int nl = lane & 15, mh = (lane >> 4) << 3;
    float pb = proj_b[nt * 16 + nl];
    #pragma unroll
    for (int r = 0; r < 8; ++r)
      yo[(mt * 16 + mh + r) * DIM + nt * 16 + nl] = acc[r] + pb;
  }
}

extern "C" void kernel_launch(void* const* d_in, const int* in_sizes, int n_in,
                              void* d_out, int out_size, void* d_ws, size_t ws_size,
                              hipStream_t stream) {
  const float* x          = (const float*)d_in[0];
  const float* mask       = (const float*)d_in[1];
  const float* qkv_w      = (const float*)d_in[2];
  const float* qkv_b      = (const float*)d_in[3];
  const float* proj_w     = (const float*)d_in[4];
  const float* proj_b     = (const float*)d_in[5];
  const float* bias_table = (const float*)d_in[6];
  const int*   rel_index  = (const int*)d_in[7];

  char* ws = (char*)d_ws;
  float* rpb  = (float*)ws;                      // 6*64*64*4   = 98304 B
  u16*  WqkvB = (u16*)(ws + 98304);              // 192*576*2   = 221184 B
  u16*  ProjB = (u16*)(ws + 98304 + 221184);     // 192*192*2   = 73728 B

  prep_rpb_kernel<<<(NH * NTOK * NTOK + 255) / 256, 256, 0, stream>>>(bias_table, rel_index, rpb);
  prep_w_kernel<<<(192 * 576 + 255) / 256, 256, 0, stream>>>(qkv_w, WqkvB, 192, 576);
  prep_w_kernel<<<(192 * 192 + 255) / 256, 256, 0, stream>>>(proj_w, ProjB, 192, 192);

  winattn_kernel<<<8192, 256, 0, stream>>>(x, mask, WqkvB, qkv_b, ProjB, proj_b, rpb, (float*)d_out);
}
